// GraphVAEEncoder_41635412967592
// MI455X (gfx1250) — compile-verified, matrix operand order checked by reading, not run on hardware
//
#include <hip/hip_runtime.h>
#include <hip/hip_bf16.h>

// ---------------------------------------------------------------------------
// Types for CDNA5 WMMA (wave32, V_WMMA_F32_16X16X32_BF16)
// ---------------------------------------------------------------------------
typedef __bf16 bf16_t;
typedef __attribute__((ext_vector_type(16))) __bf16 v16bf;
typedef __attribute__((ext_vector_type(8)))  __bf16 v8bf;
typedef __attribute__((ext_vector_type(4)))  __bf16 v4bf;
typedef __attribute__((ext_vector_type(8)))  float  v8f;
typedef __attribute__((ext_vector_type(4)))  int    v4i;

// Pointer types matching the async-to-LDS builtin signature:
// param0: int4 in addrspace(1) ("__device__"), param1: int4 in addrspace(3).
typedef __attribute__((address_space(1))) v4i* gv4i_p;
typedef __attribute__((address_space(3))) v4i* lv4i_p;

// LDS row stride (bf16 elems) padded 32 -> 40: 16 consecutive rows accessed
// with 16B ds_load_b128 start at banks (20*n)%64 -> conflict-free.
#define LDS_STRIDE 40

// CDNA5 async global->LDS path (ASYNCcnt-tracked DMA), guarded so the file
// compiles on toolchains without the builtin (falls back to sync copies).
#if defined(__HIP_DEVICE_COMPILE__) && __has_builtin(__builtin_amdgcn_global_load_async_to_lds_b128)
#define USE_ASYNC_TILES 1
#if __has_builtin(__builtin_amdgcn_s_wait_asynccnt)
#define WAIT_ASYNC(n) __builtin_amdgcn_s_wait_asynccnt(n)
#else
#define WAIT_ASYNC(n) asm volatile("s_wait_asynccnt %0" ::"n"(n) : "memory")
#endif
#else
#define USE_ASYNC_TILES 0
#endif

// Fragment loader: CDNA5 16-bit A/B layout -> lane L holds row (L&15),
// K-half (L>>4); elements 0..7 at K = kh*8, elements 8..15 at K = 16+kh*8:
// two contiguous 16-byte LDS loads.
__device__ __forceinline__ v16bf load_frag16(const bf16_t* rowbase, int khalf) {
    union { v16bf v; v8bf h[2]; } u;
    u.h[0] = *(const v8bf*)(rowbase + khalf * 8);
    u.h[1] = *(const v8bf*)(rowbase + 16 + khalf * 8);
    return u.v;
}

// ---------------------------------------------------------------------------
// Tiled GEMM  C[M,N] = A[M,K] * Bt[N,K]^T  (+bias, optional relu)
// A: bf16 row-major [M][K];  Bt: bf16 row-major [N][K] (pre-transposed W).
// Block tile 128(M) x 64(N), K-step 32, 256 threads = 8 waves; wave w owns
// rows [w*16, w*16+16) across 4 column sub-tiles. Requires K%32==0, N%64==0.
// Out-of-range M rows are CLAMPED on load (never stored) -> no divergence.
// ---------------------------------------------------------------------------
__global__ __launch_bounds__(256, 2)
void gemm_bf16_wmma(const bf16_t* __restrict__ A, const bf16_t* __restrict__ Bt,
                    float* __restrict__ C, const float* __restrict__ bias,
                    int M, int N, int K, int relu) {
    const int tid   = threadIdx.x;
    const int lane  = tid & 31;
    const int w     = tid >> 5;
    const int nl    = lane & 15;
    const int khalf = lane >> 4;
    const int blockM = blockIdx.x * 128;
    const int blockN = blockIdx.y * 64;

    v8f acc0 = {}, acc1 = {}, acc2 = {}, acc3 = {};

    auto mma_step = [&](const bf16_t* bufA, const bf16_t* bufB) {
        v16bf a  = load_frag16(bufA + (w * 16 + nl) * LDS_STRIDE, khalf);
        v16bf b0 = load_frag16(bufB + ( 0 + nl) * LDS_STRIDE, khalf);
        v16bf b1 = load_frag16(bufB + (16 + nl) * LDS_STRIDE, khalf);
        v16bf b2 = load_frag16(bufB + (32 + nl) * LDS_STRIDE, khalf);
        v16bf b3 = load_frag16(bufB + (48 + nl) * LDS_STRIDE, khalf);
        acc0 = __builtin_amdgcn_wmma_f32_16x16x32_bf16(false, a, false, b0, (short)0, acc0, false, false);
        acc1 = __builtin_amdgcn_wmma_f32_16x16x32_bf16(false, a, false, b1, (short)0, acc1, false, false);
        acc2 = __builtin_amdgcn_wmma_f32_16x16x32_bf16(false, a, false, b2, (short)0, acc2, false, false);
        acc3 = __builtin_amdgcn_wmma_f32_16x16x32_bf16(false, a, false, b3, (short)0, acc3, false, false);
    };

#if USE_ASYNC_TILES
    // Double-buffered tiles filled by GLOBAL_LOAD_ASYNC_TO_LDS_B128:
    // next tile's DMA overlaps current tile's WMMA.
    __shared__ alignas(16) bf16_t sA[2][128 * LDS_STRIDE];
    __shared__ alignas(16) bf16_t sB[2][64 * LDS_STRIDE];

    auto stage = [&](int kt, int buf) {
        // A tile: 128 rows x 32 k = 512 16B chunks, 2 per thread
        #pragma unroll
        for (int i = 0; i < 2; ++i) {
            int q = tid + i * 256;
            int r = q >> 2, c = (q & 3) << 3;
            int grow = blockM + r; if (grow >= M) grow = M - 1;   // clamp
            __builtin_amdgcn_global_load_async_to_lds_b128(
                (gv4i_p)(void*)(A + (size_t)grow * K + kt + c),
                (lv4i_p)(void*)(&sA[buf][r * LDS_STRIDE + c]),
                0, 0);
        }
        // B tile: 64 rows x 32 k = 256 16B chunks, 1 per thread
        {
            int nq = tid >> 2, c = (tid & 3) << 3;
            __builtin_amdgcn_global_load_async_to_lds_b128(
                (gv4i_p)(void*)(Bt + (size_t)(blockN + nq) * K + kt + c),
                (lv4i_p)(void*)(&sB[buf][nq * LDS_STRIDE + c]),
                0, 0);
        }
    };

    const int NT = K >> 5;
    stage(0, 0);
    for (int t = 0; t < NT; ++t) {
        int cur = t & 1;
        if (t + 1 < NT) { stage((t + 1) << 5, cur ^ 1); WAIT_ASYNC(3); }
        else            { WAIT_ASYNC(0); }
        __syncthreads();          // current tile visible to all waves
        mma_step(sA[cur], sB[cur]);
        __syncthreads();          // all reads done before buffer reuse
    }
#else
    // Synchronous fallback: single-buffered 16B copies.
    __shared__ alignas(16) bf16_t sA1[128 * LDS_STRIDE];
    __shared__ alignas(16) bf16_t sB1[64 * LDS_STRIDE];
    for (int kt = 0; kt < K; kt += 32) {
        __syncthreads();
        #pragma unroll
        for (int i = 0; i < 2; ++i) {
            int q = tid + i * 256;
            int r = q >> 2, c = (q & 3) << 3;
            int grow = blockM + r; if (grow >= M) grow = M - 1;
            v8bf v = *(const v8bf*)(A + (size_t)grow * K + kt + c);
            *(v8bf*)(&sA1[r * LDS_STRIDE + c]) = v;
        }
        {
            int nq = tid >> 2, c = (tid & 3) << 3;
            v8bf v = *(const v8bf*)(Bt + (size_t)(blockN + nq) * K + kt + c);
            *(v8bf*)(&sB1[nq * LDS_STRIDE + c]) = v;
        }
        __syncthreads();
        mma_step(sA1, sB1);
    }
#endif

    // Epilogue: C/D layout -> lane holds N=lane&15, VGPR v holds
    // M = v + 8*(lane>>4) within the 16x16 tile.
    const int rbase = blockM + w * 16 + khalf * 8;
    v8f accs[4] = {acc0, acc1, acc2, acc3};
    #pragma unroll
    for (int j = 0; j < 4; ++j) {
        int col = blockN + j * 16 + nl;
        float bv = bias ? bias[col] : 0.f;
        #pragma unroll
        for (int v = 0; v < 8; ++v) {
            int row = rbase + v;
            if (row < M) {
                float x = accs[j][v] + bv;
                if (relu) x = fmaxf(x, 0.f);
                C[(size_t)row * N + col] = x;
            }
        }
    }
}

// ---------------------------------------------------------------------------
// Precision / layout conversion kernels
// ---------------------------------------------------------------------------
__global__ void k_cvt_bf16(const float* __restrict__ in, bf16_t* __restrict__ out, int total4) {
    int q = blockIdx.x * blockDim.x + threadIdx.x;
    if (q >= total4) return;
    float4 v = *(const float4*)(in + (size_t)q * 4);
    v4bf o = {(bf16_t)v.x, (bf16_t)v.y, (bf16_t)v.z, (bf16_t)v.w};
    *(v4bf*)(out + (size_t)q * 4) = o;
}

// Bt[n*K + k] = (bf16) W[k*N + n]   (weight transpose + convert; tiny)
__global__ void k_wt_bf16(const float* __restrict__ W, bf16_t* __restrict__ Bt, int K, int N) {
    int idx = blockIdx.x * blockDim.x + threadIdx.x;
    if (idx >= N * K) return;
    int n = idx / K, k = idx - n * K;
    Bt[idx] = (bf16_t)W[(size_t)k * N + n];
}

// relu + convert: activations are only consumed by later bf16 GEMMs
__global__ void k_relu_cvt(const float* __restrict__ in, bf16_t* __restrict__ out, int total4) {
    int q = blockIdx.x * blockDim.x + threadIdx.x;
    if (q >= total4) return;
    float4 v = *(const float4*)(in + (size_t)q * 4);
    v4bf o = {(bf16_t)fmaxf(v.x, 0.f), (bf16_t)fmaxf(v.y, 0.f),
              (bf16_t)fmaxf(v.z, 0.f), (bf16_t)fmaxf(v.w, 0.f)};
    *(v4bf*)(out + (size_t)q * 4) = o;
}

// ---------------------------------------------------------------------------
// Degree / normalization kernels
// ---------------------------------------------------------------------------
__global__ void k_deg_init(float* __restrict__ deg, int n) {
    int i = blockIdx.x * blockDim.x + threadIdx.x;
    if (i < n) deg[i] = 1.0f;                       // self-loop weight
}

__global__ void k_deg_accum(const int* __restrict__ dst, const float* __restrict__ ew,
                            float* __restrict__ deg, int E) {
    int e = blockIdx.x * blockDim.x + threadIdx.x;
    if (e < E) atomicAdd(deg + dst[e], ew[e]);
}

__global__ void k_dis(const float* __restrict__ deg, float* __restrict__ dis, int n) {
    int i = blockIdx.x * blockDim.x + threadIdx.x;
    if (i < n) dis[i] = rsqrtf(deg[i]);             // deg >= 1 always
}

__global__ void k_norm(const int* __restrict__ src, const int* __restrict__ dst,
                       const float* __restrict__ ew, const float* __restrict__ dis,
                       float* __restrict__ norm, int E) {
    int e = blockIdx.x * blockDim.x + threadIdx.x;
    if (e < E) norm[e] = dis[src[e]] * ew[e] * dis[dst[e]];
}

// out[i,:] = dis[i]^2 * t[i,:] + b[:]   (self-loop message + bias)
__global__ void k_agg_init(const float* __restrict__ t, const float* __restrict__ dis,
                           const float* __restrict__ b, float* __restrict__ out,
                           int n, int F) {
    int q = blockIdx.x * blockDim.x + threadIdx.x;          // in float4 units
    int total = n * (F >> 2);
    if (q >= total) return;
    int i = q / (F >> 2);
    int f = (q - i * (F >> 2)) << 2;
    float s = dis[i] * dis[i];
    float4 tv = *(const float4*)(t + (size_t)i * F + f);
    float4 bv = *(const float4*)(b + f);
    float4 o;
    o.x = s * tv.x + bv.x; o.y = s * tv.y + bv.y;
    o.z = s * tv.z + bv.z; o.w = s * tv.w + bv.w;
    *(float4*)(out + (size_t)i * F + f) = o;
}

// one block per edge: out[dst,:] += norm[e] * t[src,:]   (t, out L2-resident)
__global__ __launch_bounds__(128)
void k_scatter(const float* __restrict__ t, const float* __restrict__ norm,
               const int* __restrict__ src, const int* __restrict__ dst,
               float* __restrict__ out, int F) {
    int e = blockIdx.x;
    int s = src[e], d = dst[e];
    float wgt = norm[e];
    int f = threadIdx.x << 2;                       // 128 threads * 4 = 512 feats
    float4 v = *(const float4*)(t + (size_t)s * F + f);
    float* o = out + (size_t)d * F + f;
    atomicAdd(o + 0, wgt * v.x);
    atomicAdd(o + 1, wgt * v.y);
    atomicAdd(o + 2, wgt * v.z);
    atomicAdd(o + 3, wgt * v.w);
}

// ---------------------------------------------------------------------------
// Launch
// ---------------------------------------------------------------------------
extern "C" void kernel_launch(void* const* d_in, const int* in_sizes, int n_in,
                              void* d_out, int out_size, void* d_ws, size_t ws_size,
                              hipStream_t stream) {
    const int DIN = 512, DH = 512, DZ = 64;
    const float* x   = (const float*)d_in[0];
    const int*   ei  = (const int*)d_in[1];
    const float* ew  = (const float*)d_in[2];
    const float* W1  = (const float*)d_in[3];
    const float* b1  = (const float*)d_in[4];
    const float* W2  = (const float*)d_in[5];
    const float* b2  = (const float*)d_in[6];
    const float* Wmu = (const float*)d_in[7];
    const float* bmu = (const float*)d_in[8];
    const float* Wlv = (const float*)d_in[9];
    const float* blv = (const float*)d_in[10];

    const int n = in_sizes[0] / DIN;       // 50000
    const int E = in_sizes[2];             // 1.6M
    const int* src = ei;
    const int* dst = ei + E;

    // workspace layout
    char* p = (char*)d_ws;
    float*  t    = (float*)p;  p += (size_t)n * DH * 4;     // GEMM out (f32)
    float*  h    = (float*)p;  p += (size_t)n * DH * 4;     // aggregated (f32)
    bf16_t* xb   = (bf16_t*)p; p += (size_t)n * DIN * 2;    // bf16 input copy
    bf16_t* hb   = (bf16_t*)p; p += (size_t)n * DH * 2;     // bf16 relu acts
    bf16_t* W1t  = (bf16_t*)p; p += (size_t)DIN * DH * 2;   // transposed bf16 W
    bf16_t* W2t  = (bf16_t*)p; p += (size_t)DH * DH * 2;
    bf16_t* Wmt  = (bf16_t*)p; p += (size_t)DH * DZ * 2;
    bf16_t* Wlt  = (bf16_t*)p; p += (size_t)DH * DZ * 2;
    float*  deg  = (float*)p;  p += (size_t)n * 4;
    float*  dis  = (float*)p;  p += (size_t)n * 4;
    float*  nrm  = (float*)p;

    float* mu = (float*)d_out;
    float* lv = mu + (size_t)n * DZ;

    const int TB = 256;
    dim3 gGemmL((n + 127) / 128, DH / 64);  // layer GEMMs: N=512
    dim3 gGemmH((n + 127) / 128, DZ / 64);  // head GEMMs:  N=64
    int nf4 = n * (DH / 4);

    // 0) precision/layout prep
    k_cvt_bf16<<<(nf4 + TB - 1) / TB, TB, 0, stream>>>(x, xb, nf4);
    k_wt_bf16<<<(DIN * DH + TB - 1) / TB, TB, 0, stream>>>(W1, W1t, DIN, DH);
    k_wt_bf16<<<(DH * DH + TB - 1) / TB, TB, 0, stream>>>(W2, W2t, DH, DH);
    k_wt_bf16<<<(DH * DZ + TB - 1) / TB, TB, 0, stream>>>(Wmu, Wmt, DH, DZ);
    k_wt_bf16<<<(DH * DZ + TB - 1) / TB, TB, 0, stream>>>(Wlv, Wlt, DH, DZ);

    // 1) normalization coefficients
    k_deg_init<<<(n + TB - 1) / TB, TB, 0, stream>>>(deg, n);
    k_deg_accum<<<(E + TB - 1) / TB, TB, 0, stream>>>(dst, ew, deg, E);
    k_dis<<<(n + TB - 1) / TB, TB, 0, stream>>>(deg, dis, n);
    k_norm<<<(E + TB - 1) / TB, TB, 0, stream>>>(src, dst, ew, dis, nrm, E);

    // 2) layer 1: t = x @ W1 ; h = scatter + self + b1 ; hb = bf16(relu(h))
    gemm_bf16_wmma<<<gGemmL, TB, 0, stream>>>(xb, W1t, t, nullptr, n, DH, DIN, 0);
    k_agg_init<<<(nf4 + TB - 1) / TB, TB, 0, stream>>>(t, dis, b1, h, n, DH);
    k_scatter<<<E, 128, 0, stream>>>(t, nrm, src, dst, h, DH);
    k_relu_cvt<<<(nf4 + TB - 1) / TB, TB, 0, stream>>>(h, hb, nf4);

    // 3) layer 2
    gemm_bf16_wmma<<<gGemmL, TB, 0, stream>>>(hb, W2t, t, nullptr, n, DH, DH, 0);
    k_agg_init<<<(nf4 + TB - 1) / TB, TB, 0, stream>>>(t, dis, b2, h, n, DH);
    k_scatter<<<E, 128, 0, stream>>>(t, nrm, src, dst, h, DH);
    k_relu_cvt<<<(nf4 + TB - 1) / TB, TB, 0, stream>>>(h, hb, nf4);

    // 4) heads straight into d_out: mu ‖ logvar
    gemm_bf16_wmma<<<gGemmH, TB, 0, stream>>>(hb, Wmt, mu, bmu, n, DZ, DH, 0);
    gemm_bf16_wmma<<<gGemmH, TB, 0, stream>>>(hb, Wlt, lv, blv, n, DZ, DH, 0);
}
